// Generator_15487652069776
// MI455X (gfx1250) — compile-verified, matrix operand order checked by reading, not run on hardware
//
#include <hip/hip_runtime.h>
#include <hip/hip_bf16.h>
#include <hip/hip_fp16.h>

// ---------------------------------------------------------------------------
// Problem constants (match reference)
// ---------------------------------------------------------------------------
#define Bsz  32
#define Tlen 256
#define Vsz  32000
#define Hdim 512
#define Lnum 2
#define NBLK 32          // persistent blocks in GRU scan (one per 16 h-columns)

typedef __attribute__((ext_vector_type(16))) _Float16 v16h;
typedef __attribute__((ext_vector_type(8)))  _Float16 v8h;
typedef __attribute__((ext_vector_type(8)))  float    v8f;

// ---------------------------------------------------------------------------
// WMMA helpers
// ---------------------------------------------------------------------------
// Load one 16x32 f16 operand tile (A layout; B uses the symmetric layout with
// "row" = weight output-neuron row, since gi = x * W^T means B[k][n] = W[n][k]
// and W's natural [N][K] row-major storage is B column-major).
// ISA 16-bit A 16x32 layout: lanes 0-15 hold K {0..7, 16..23} for row M=lane,
// lanes 16-31 hold K {8..15, 24..31} for row M=lane-16  -> two b128 loads.
__device__ __forceinline__ v16h load_tile16(const _Float16* __restrict__ base,
                                            int ld, int row0, int k0, int lane) {
    const int r  = lane & 15;
    const int kh = lane >> 4;
    const _Float16* p = base + (size_t)(row0 + r) * ld + k0 + kh * 8;
    union { v16h v; v8h h[2]; } u;
    u.h[0] = *(const v8h*)(p);       // K +0..7   (relative to kh*8)
    u.h[1] = *(const v8h*)(p + 16);  // K +16..23
    return u.v;
}

__device__ __forceinline__ v8f wmma_f16(v16h a, v16h b, v8f c) {
    // D = A(16x32) * B(32x16) + C, f32 accumulate
    return __builtin_amdgcn_wmma_f32_16x16x32_f16(
        /*neg_a=*/false, a, /*neg_b=*/false, b,
        /*c_mod=*/(short)0, c, /*reuse_a=*/false, /*reuse_b=*/false);
}

// ---------------------------------------------------------------------------
// Utility kernels
// ---------------------------------------------------------------------------
__global__ void k_init(_Float16* __restrict__ hbuf, unsigned* __restrict__ bar) {
    int i = blockIdx.x * blockDim.x + threadIdx.x;
    if (i < 2 * Bsz * Hdim) hbuf[i] = (_Float16)0.f;   // h0 = zeros (both buffers)
    if (i < 2)              bar[i]  = 0u;              // barrier count + sense
}

__global__ void k_cvt(_Float16* __restrict__ dst, const float* __restrict__ src, int n) {
    int i = blockIdx.x * blockDim.x + threadIdx.x;
    if (i < n) dst[i] = (_Float16)src[i];
}

// E16[(t*B + b)*H + h] = emb[x[b*T + t]][h]  (fp16)
__global__ void k_embed(_Float16* __restrict__ E16, const float* __restrict__ emb,
                        const int* __restrict__ x, int n) {
    int i = blockIdx.x * blockDim.x + threadIdx.x;
    if (i >= n) return;
    int hcol = i & (Hdim - 1);
    int tb   = i >> 9;          // / Hdim
    int t    = tb >> 5;         // / Bsz
    int b    = tb & 31;
    int tok  = x[b * Tlen + t];
    E16[i] = (_Float16)emb[(size_t)tok * Hdim + hcol];
}

// ---------------------------------------------------------------------------
// Persistent GRU scan: grid = 32 blocks x 256 threads. Block bx owns h columns
// [bx*16, bx*16+16). 8 waves = 2 M-tiles x 4 roles {r(fused), z(fused), inn, hn}.
// Ping-pong hidden buffer + device-scope sense barrier between the 512 phases.
// ---------------------------------------------------------------------------
__global__ void __launch_bounds__(256) k_gru_scan(
    _Float16* __restrict__ hbuf,        // [2][B][H] fp16 ping-pong hidden
    const _Float16* __restrict__ E16,   // [T][B][H]
    const _Float16* __restrict__ Wih16, // [L][3H][H]
    const _Float16* __restrict__ Whh16, // [L][3H][H]
    const float*    __restrict__ bih,   // [L][3H]
    const float*    __restrict__ bhh,   // [L][3H]
    _Float16* __restrict__ outs16,      // [T][B][H] (hidden after layer 2)
    unsigned* __restrict__ bar)         // [0]=count, [1]=sense/generation
{
    __shared__ float lds[4][2][16][16]; // [role][mtile][row][col]
    const int tid  = threadIdx.x;
    const int lane = tid & 31;
    const int wave = tid >> 5;
    const int mt   = wave & 1;          // M tile (batch rows 0-15 / 16-31)
    const int role = wave >> 1;         // 0=r, 1=z, 2=inn, 3=hn
    const int n0   = blockIdx.x * 16;   // owned h-column range
    const int BH   = Bsz * Hdim;

    for (int phase = 0; phase < 2 * Tlen; ++phase) {
        const int t = phase >> 1;
        const int l = phase & 1;
        const _Float16* hrd = hbuf + (size_t)(phase & 1) * BH;
        _Float16*       hwr = hbuf + (size_t)((phase + 1) & 1) * BH;
        // Layer-threading quirk: layer 1's input IS the current hidden.
        const _Float16* xp = l ? hrd : (E16 + (size_t)t * BH);
        const _Float16* Wi = Wih16 + (size_t)l * 3 * Hdim * Hdim;
        const _Float16* Wh = Whh16 + (size_t)l * 3 * Hdim * Hdim;
        const int g = (role < 2) ? role : 2;

        v8f acc = {};
        if (role != 3) {                         // x * Wih_g^T contribution
            #pragma unroll 4
            for (int k0 = 0; k0 < Hdim; k0 += 32) {
                v16h a = load_tile16(xp, Hdim, mt * 16, k0, lane);
                v16h b = load_tile16(Wi, Hdim, g * Hdim + n0, k0, lane);
                acc = wmma_f16(a, b, acc);
            }
        }
        if (role != 2) {                         // h * Whh_g^T contribution
            #pragma unroll 4
            for (int k0 = 0; k0 < Hdim; k0 += 32) {
                v16h a = load_tile16(hrd, Hdim, mt * 16, k0, lane);
                v16h b = load_tile16(Wh, Hdim, g * Hdim + n0, k0, lane);
                acc = wmma_f16(a, b, acc);
            }
        }
        { // spill accumulator tile to LDS (C/D layout: col=lane%16, rows 8*(lane/16)+i)
            const int col = lane & 15, rb = (lane >> 4) * 8;
            #pragma unroll
            for (int i = 0; i < 8; ++i) lds[role][mt][rb + i][col] = acc[i];
        }
        __syncthreads();

        // Elementwise gate math: 2*16*16 = 512 h-values, 2 per thread.
        #pragma unroll
        for (int p = 0; p < 2; ++p) {
            const int f   = tid + p * 256;
            const int m2  = f >> 8;
            const int row = (f >> 4) & 15;
            const int col = f & 15;
            const int b   = m2 * 16 + row;
            const int j   = n0 + col;
            const int bb  = l * 3 * Hdim + j;
            float rg  = lds[0][m2][row][col] + bih[bb]            + bhh[bb];
            float zg  = lds[1][m2][row][col] + bih[bb + Hdim]     + bhh[bb + Hdim];
            float inn = lds[2][m2][row][col] + bih[bb + 2 * Hdim];
            float hn  = lds[3][m2][row][col] + bhh[bb + 2 * Hdim];
            float r = 1.f / (1.f + __expf(-rg));
            float z = 1.f / (1.f + __expf(-zg));
            float n = tanhf(inn + r * hn);
            float hold = (float)hrd[b * Hdim + j];
            float hnew = (1.f - z) * n + z * hold;
            hwr[b * Hdim + j] = (_Float16)hnew;
            if (l) outs16[((size_t)t * Bsz + b) * Hdim + j] = (_Float16)hnew;
        }
        __threadfence();

        // Device-wide sense-reversal barrier.
        __syncthreads();
        if (tid == 0) {
            unsigned my  = __hip_atomic_load(&bar[1], __ATOMIC_ACQUIRE, __HIP_MEMORY_SCOPE_AGENT);
            unsigned old = __hip_atomic_fetch_add(&bar[0], 1u, __ATOMIC_ACQ_REL, __HIP_MEMORY_SCOPE_AGENT);
            if (old == NBLK - 1) {
                __hip_atomic_store(&bar[0], 0u, __ATOMIC_RELAXED, __HIP_MEMORY_SCOPE_AGENT);
                __hip_atomic_fetch_add(&bar[1], 1u, __ATOMIC_RELEASE, __HIP_MEMORY_SCOPE_AGENT);
            } else {
                while (__hip_atomic_load(&bar[1], __ATOMIC_ACQUIRE, __HIP_MEMORY_SCOPE_AGENT) == my)
                    __builtin_amdgcn_s_sleep(2);
            }
        }
        __syncthreads();
    }
}

// ---------------------------------------------------------------------------
// Output projection: logits[b,t,v] = outs[tb,:] . Wout[v,:] + bout[v]
// A = outs16 [8192][512], B = Wout16 [32000][512] (row = B column).
// Block tile 32(M) x 256(N): 8 waves = 2 M-tiles x 4 N-quads. Each wave keeps
// ONE A fragment and FOUR accumulators (1 M-tile x 4 N-tiles): 4 WMMAs per A
// load -> 10 b128 loads per 4 WMMAs instead of 16 (4x A reuse in registers).
// ---------------------------------------------------------------------------
__global__ void __launch_bounds__(256) k_out_gemm(
    float* __restrict__ out, const _Float16* __restrict__ A,
    const _Float16* __restrict__ Wout16, const float* __restrict__ bout)
{
    const int lane = threadIdx.x & 31;
    const int wave = threadIdx.x >> 5;
    const int mi   = wave & 1;                     // 2 M-tiles
    const int nq   = wave >> 1;                    // 4 N-quads
    const int row0 = blockIdx.y * 32 + mi * 16;    // tb rows
    const int nb   = blockIdx.x * 256 + nq * 64;   // vocab col base (4 tiles)

    v8f acc0 = {}, acc1 = {}, acc2 = {}, acc3 = {};
    #pragma unroll 2
    for (int k0 = 0; k0 < Hdim; k0 += 32) {
        v16h a  = load_tile16(A, Hdim, row0, k0, lane);
        v16h b0 = load_tile16(Wout16, Hdim, nb,      k0, lane);
        v16h b1 = load_tile16(Wout16, Hdim, nb + 16, k0, lane);
        v16h b2 = load_tile16(Wout16, Hdim, nb + 32, k0, lane);
        v16h b3 = load_tile16(Wout16, Hdim, nb + 48, k0, lane);
        acc0 = wmma_f16(a, b0, acc0);
        acc1 = wmma_f16(a, b1, acc1);
        acc2 = wmma_f16(a, b2, acc2);
        acc3 = wmma_f16(a, b3, acc3);
    }

    const int col = lane & 15;
    const int rb  = (lane >> 4) * 8;
    v8f accs[4] = {acc0, acc1, acc2, acc3};
    #pragma unroll
    for (int j = 0; j < 4; ++j) {
        const int v  = nb + 16 * j + col;
        const float bv = bout[v];
        #pragma unroll
        for (int i = 0; i < 8; ++i) {
            const int tb = row0 + rb + i;
            const int t  = tb >> 5;
            const int b  = tb & 31;
            out[(size_t)b * ((size_t)Tlen * Vsz) + (size_t)t * Vsz + v] = accs[j][i] + bv;
        }
    }
}

// ---------------------------------------------------------------------------
// Launch
// ---------------------------------------------------------------------------
extern "C" void kernel_launch(void* const* d_in, const int* in_sizes, int n_in,
                              void* d_out, int out_size, void* d_ws, size_t ws_size,
                              hipStream_t stream) {
    const int*   x    = (const int*)  d_in[0];
    const float* emb  = (const float*)d_in[1];
    const float* Wih  = (const float*)d_in[2];
    const float* Whh  = (const float*)d_in[3];
    const float* bih  = (const float*)d_in[4];
    const float* bhh  = (const float*)d_in[5];
    const float* Wout = (const float*)d_in[6];
    const float* bout = (const float*)d_in[7];
    float* out = (float*)d_out;

    // Workspace carving (256B aligned)
    char* ws = (char*)d_ws;
    size_t off = 0;
    auto carve = [&](size_t bytes) { char* p = ws + off; off = (off + bytes + 255) & ~(size_t)255; return p; };
    unsigned*  bar    = (unsigned*) carve(256);
    _Float16*  hbuf   = (_Float16*) carve((size_t)2 * Bsz * Hdim * 2);
    _Float16*  E16    = (_Float16*) carve((size_t)Tlen * Bsz * Hdim * 2);
    _Float16*  outs16 = (_Float16*) carve((size_t)Tlen * Bsz * Hdim * 2);
    _Float16*  Wih16  = (_Float16*) carve((size_t)Lnum * 3 * Hdim * Hdim * 2);
    _Float16*  Whh16  = (_Float16*) carve((size_t)Lnum * 3 * Hdim * Hdim * 2);
    _Float16*  Wout16 = (_Float16*) carve((size_t)Vsz * Hdim * 2);
    (void)ws_size; (void)n_in; (void)in_sizes; (void)out_size;

    // 1) zero hidden ping-pong + barrier state
    k_init<<<(2 * Bsz * Hdim + 255) / 256, 256, 0, stream>>>(hbuf, bar);

    // 2) embedding gather -> fp16 [T][B][H]
    {
        int n = Tlen * Bsz * Hdim;
        k_embed<<<(n + 255) / 256, 256, 0, stream>>>(E16, emb, x, n);
    }

    // 3) fp16 weight conversions (L2-resident thereafter)
    {
        int nW = Lnum * 3 * Hdim * Hdim;
        k_cvt<<<(nW + 255) / 256, 256, 0, stream>>>(Wih16, Wih, nW);
        k_cvt<<<(nW + 255) / 256, 256, 0, stream>>>(Whh16, Whh, nW);
        int nO = Vsz * Hdim;
        k_cvt<<<(nO + 255) / 256, 256, 0, stream>>>(Wout16, Wout, nO);
    }

    // 4) persistent recurrent scan (512 phases, grid barrier per phase)
    k_gru_scan<<<NBLK, 256, 0, stream>>>(hbuf, E16, Wih16, Whh16, bih, bhh, outs16, bar);

    // 5) output projection GEMM [8192x512]x[512x32000] -> logits [B][T][V]
    {
        dim3 grid(Vsz / 256, (Tlen * Bsz) / 32);
        k_out_gemm<<<grid, 256, 0, stream>>>(out, outs16, Wout16, bout);
    }
}
